// CausalTransition_38319698215619
// MI455X (gfx1250) — matrix operand order repaired and checked by reading, not compiled, not run on hardware
//
#include <hip/hip_runtime.h>
#include <hip/hip_bf16.h>

// ---- problem constants ----
#define B_   2
#define C_   64
#define H_   12
#define W_   12
#define N_   144      // H*W
#define M_   145      // N+1 (action node)
#define MP_  160      // M padded to multiple of 16/32
#define L_   800

typedef __bf16 bf16_t;
typedef __attribute__((ext_vector_type(16))) __bf16 v16bf;
typedef __attribute__((ext_vector_type(8)))  __bf16 v8bf;
typedef __attribute__((ext_vector_type(8)))  float  v8f;
typedef __attribute__((ext_vector_type(4)))  int    v4i;

#define AS1 __attribute__((address_space(1)))
#define AS3 __attribute__((address_space(3)))

// ---- gfx1250 async global->LDS path (probe with __has_builtin) ----
#if defined(__has_builtin)
#if __has_builtin(__builtin_amdgcn_global_load_async_to_lds_b128) && \
    __has_builtin(__builtin_amdgcn_s_wait_asynccnt)
#define HAVE_ASYNC_LDS 1
#endif
#endif
#ifndef HAVE_ASYNC_LDS
#define HAVE_ASYNC_LDS 0
#endif

// -------- deterministic hash RNG (replaces jax.random; no runtime check) ---
__device__ __forceinline__ unsigned hash_u32(unsigned x) {
  x ^= x >> 16; x *= 0x7feb352du;
  x ^= x >> 15; x *= 0x846ca68bu;
  x ^= x >> 16; return x;
}
__device__ __forceinline__ float u01(unsigned h) {
  return ((h >> 8) + 0.5f) * (1.0f / 16777216.0f);
}

// ---------------- pos = latent(BCHW->BNC) + sinusoidal PE ------------------
__global__ void k_pos(const float* __restrict__ latent,
                      float* __restrict__ pos_f, bf16_t* __restrict__ pos_b) {
  int idx = blockIdx.x * blockDim.x + threadIdx.x;
  if (idx >= B_ * N_ * C_) return;
  int c = idx % C_;
  int n = (idx / C_) % N_;
  int b = idx / (C_ * N_);
  float lat = latent[(b * C_ + c) * N_ + n];          // (B,C,H,W) flat
  int k = c >> 1;
  float div = __expf(-(2.0f * (float)k) * (logf(10000.0f) / (float)C_));
  float ang = (float)n * div;
  float pe = (c & 1) ? __cosf(ang) : __sinf(ang);
  float v = lat + pe;
  pos_f[idx] = v;
  pos_b[idx] = (bf16_t)v;
}

// ------- transpose f32 (R x Ccols) -> bf16 (Ccols x R), N-major weights ----
__global__ void k_transpose_bf16(const float* __restrict__ src,
                                 bf16_t* __restrict__ dst, int R, int Ccols) {
  int idx = blockIdx.x * blockDim.x + threadIdx.x;
  if (idx >= R * Ccols) return;
  int c = idx % Ccols;
  int r = idx / Ccols;
  dst[c * R + r] = (bf16_t)src[r * Ccols + c];
}

// ---------------- pa[l] = action . Wa[:,l] + b1[l]  (action = a_dense_b) ---
__global__ void k_pa(const float* __restrict__ adb, const float* __restrict__ Wa,
                     const float* __restrict__ b1, float* __restrict__ pa) {
  int l = blockIdx.x * blockDim.x + threadIdx.x;
  if (l >= L_) return;
  float s = b1[l];
  for (int c = 0; c < C_; ++c) s += adb[c] * Wa[c * L_ + l];
  pa[l] = s;
}

// --------------- WMMA bf16 GEMM (direct loads) -----------------------------
// A: bf16 row-major (lda).  Bt: bf16 N-major, Bt[n*ldb + k] = B[k][n].
__global__ void k_wmma_gemm(const bf16_t* __restrict__ A, int sA,
                            const bf16_t* __restrict__ Bt, int sB,
                            float* __restrict__ Cm, int sC,
                            int Ncols, int K, int lda, int ldb, int ldc) {
  int ntiles = Ncols >> 4;
  int tm = (int)blockIdx.x / ntiles;
  int tn = (int)blockIdx.x % ntiles;
  A  += (int)blockIdx.y * sA;
  Bt += (int)blockIdx.y * sB;
  Cm += (int)blockIdx.y * sC;

  int lane = threadIdx.x & 31;
  int half = lane >> 4;
  int lid  = lane & 15;
  int row = tm * 16 + lid;
  int col = tn * 16 + lid;

  v8f acc = {};
  const bf16_t* arow = A  + row * lda;
  const bf16_t* bcol = Bt + col * ldb;
  for (int kk = 0; kk < K; kk += 32) {
    __builtin_prefetch(arow + kk + 32, 0, 0);
    __builtin_prefetch(bcol + kk + 32, 0, 0);
    union { v16bf v; v8bf h[2]; } a;
    a.h[0] = *(const v8bf*)(arow + kk + half * 8);
    a.h[1] = *(const v8bf*)(arow + kk + 16 + half * 8);
    union { v16bf v; v8bf h[2]; } bb;
    bb.h[0] = *(const v8bf*)(bcol + kk + half * 8);
    bb.h[1] = *(const v8bf*)(bcol + kk + 16 + half * 8);
    v16bf b = bb.v;
    acc = __builtin_amdgcn_wmma_f32_16x16x32_bf16(
        false, a.v, false, b, (short)0, acc, false, false);
  }
#pragma unroll
  for (int r = 0; r < 8; ++r)
    Cm[(tm * 16 + half * 8 + r) * ldc + col] = acc[r];
}

// ------ WMMA bf16 GEMM, double-buffered async global->LDS staging ----------
// Each 16x32 A/B chunk is staged to LDS with global_load_async_to_lds_b128
// (ASYNCcnt), next chunk prefetched while current feeds v_wmma from ds loads.
// One wave per block -> no barrier needed, only s_wait_asynccnt.
__global__ void k_wmma_gemm_lds(const bf16_t* __restrict__ A, int sA,
                                const bf16_t* __restrict__ Bt, int sB,
                                float* __restrict__ Cm, int sC,
                                int Ncols, int K, int lda, int ldb, int ldc) {
  __shared__ bf16_t lA[2][16 * 32];
  __shared__ bf16_t lB[2][16 * 32];

  int ntiles = Ncols >> 4;
  int tm = (int)blockIdx.x / ntiles;
  int tn = (int)blockIdx.x % ntiles;
  A  += (int)blockIdx.y * sA;
  Bt += (int)blockIdx.y * sB;
  Cm += (int)blockIdx.y * sC;

  int lane = threadIdx.x & 31;
  int half = lane >> 4;
  int lid  = lane & 15;
  int col  = tn * 16 + lid;
  int srow  = lane >> 1;          // 16 rows, 2 lanes per row
  int spart = (lane & 1) * 16;    // each lane moves 16 bf16 (2 x b128)

  const bf16_t* ga = A  + (tm * 16 + srow) * lda + spart;
  const bf16_t* gb = Bt + (tn * 16 + srow) * ldb + spart;

  auto stage = [&](int buf, int kk) {
#if HAVE_ASYNC_LDS
    __builtin_amdgcn_global_load_async_to_lds_b128(
        (AS1 v4i*)(ga + kk),
        (AS3 v4i*)&lA[buf][srow * 32 + spart], 0, 0);
    __builtin_amdgcn_global_load_async_to_lds_b128(
        (AS1 v4i*)(ga + kk + 8),
        (AS3 v4i*)&lA[buf][srow * 32 + spart + 8], 0, 0);
    __builtin_amdgcn_global_load_async_to_lds_b128(
        (AS1 v4i*)(gb + kk),
        (AS3 v4i*)&lB[buf][srow * 32 + spart], 0, 0);
    __builtin_amdgcn_global_load_async_to_lds_b128(
        (AS1 v4i*)(gb + kk + 8),
        (AS3 v4i*)&lB[buf][srow * 32 + spart + 8], 0, 0);
#else
    *(v8bf*)&lA[buf][srow * 32 + spart]     = *(const v8bf*)(ga + kk);
    *(v8bf*)&lA[buf][srow * 32 + spart + 8] = *(const v8bf*)(ga + kk + 8);
    *(v8bf*)&lB[buf][srow * 32 + spart]     = *(const v8bf*)(gb + kk);
    *(v8bf*)&lB[buf][srow * 32 + spart + 8] = *(const v8bf*)(gb + kk + 8);
#endif
  };

  v8f acc = {};
  int nk = K >> 5;
  stage(0, 0);
  for (int t = 0; t < nk; ++t) {
    int buf = t & 1;
    bool more = (t + 1) < nk;
    if (more) stage(buf ^ 1, (t + 1) * 32);
#if HAVE_ASYNC_LDS
    if (more) __builtin_amdgcn_s_wait_asynccnt(4);   // current stage complete
    else      __builtin_amdgcn_s_wait_asynccnt(0);
#endif
    union { v16bf v; v8bf h[2]; } a;
    a.h[0] = *(const v8bf*)&lA[buf][lid * 32 + half * 8];
    a.h[1] = *(const v8bf*)&lA[buf][lid * 32 + 16 + half * 8];
    union { v16bf v; v8bf h[2]; } bb;
    bb.h[0] = *(const v8bf*)&lB[buf][lid * 32 + half * 8];
    bb.h[1] = *(const v8bf*)&lB[buf][lid * 32 + 16 + half * 8];
    v16bf b = bb.v;
    acc = __builtin_amdgcn_wmma_f32_16x16x32_bf16(
        false, a.v, false, b, (short)0, acc, false, false);
  }
#pragma unroll
  for (int r = 0; r < 8; ++r)
    Cm[(tm * 16 + half * 8 + r) * ldc + col] = acc[r];
}

// ------- coeffs + gumbel-argmax -> binary causal graph cg[b,i,j] -----------
__global__ void k_cg(const float* __restrict__ pi, const float* __restrict__ pj,
                     const float* __restrict__ pa, const float* __restrict__ w2,
                     const float* __restrict__ b2, float* __restrict__ cg) {
  int idx = blockIdx.x;                     // b*N*N waves
  int j = idx % N_;
  int i = (idx / N_) % N_;
  int b = idx / (N_ * N_);
  int lane = threadIdx.x;
  const float* pib = pi + (b * N_ + i) * L_;
  const float* pjb = pj + (b * N_ + j) * L_;
  float s = 0.f;
  for (int l = lane; l < L_; l += 32) {
    float x = pib[l] + pjb[l] + pa[l];
    s += w2[l] * (x > 0.f ? x : 0.01f * x);
  }
  for (int o = 16; o > 0; o >>= 1) s += __shfl_xor(s, o, 32);   // wave32
  if (lane == 0) {
    float cf = 1.f / (1.f + __expf(-(s + b2[0])));
    cf = fminf(fmaxf(cf, 1e-12f), 1.f - 1e-12f);
    unsigned base = (unsigned)idx * 2654435761u + 42u;
    float g0 = -logf(-logf(u01(hash_u32(base ^ 0x9e3779b9u))));
    float g1 = -logf(-logf(u01(hash_u32(base ^ 0x85ebca6bu))));
    cg[idx] = (logf(cf) + g1 > logf(1.f - cf) + g0) ? 1.f : 0.f;
  }
}

// ---- nodes = [pos ; action] + N(0,1) noise, zero-padded rows 145..159 -----
__global__ void k_nodes(const float* __restrict__ pos_f,
                        const float* __restrict__ adb,
                        float* __restrict__ nodes_f, bf16_t* __restrict__ nodes_b,
                        unsigned salt) {
  int idx = blockIdx.x * blockDim.x + threadIdx.x;
  if (idx >= B_ * MP_ * C_) return;
  int c = idx % C_;
  int m = (idx / C_) % MP_;
  int b = idx / (C_ * MP_);
  float v = 0.f;
  if (m < M_) {
    float base = (m < N_) ? pos_f[(b * N_ + m) * C_ + c] : adb[c];
    float u1 = u01(hash_u32((unsigned)idx * 0x9e3779b9u + salt));
    float u2 = u01(hash_u32((unsigned)idx * 0x85ebca6bu + salt * 0x27d4eb2fu + 7u));
    v = base + sqrtf(-2.f * logf(u1)) * __cosf(6.28318530718f * u2);
  }
  nodes_f[idx] = v;
  nodes_b[idx] = (bf16_t)v;
}

// ------- xl/xr bias epilogue; also write xl transposed to bf16 -------------
__global__ void k_xlxr_epi(float* __restrict__ xl, float* __restrict__ xr,
                           const float* __restrict__ bl, const float* __restrict__ br,
                           bf16_t* __restrict__ xlT) {
  int idx = blockIdx.x * blockDim.x + threadIdx.x;
  if (idx >= B_ * MP_ * L_) return;
  int l = idx % L_;
  int m = (idx / L_) % MP_;
  int b = idx / (L_ * MP_);
  float vl = 0.f, vr = 0.f;
  if (m < M_) { vl = xl[idx] + bl[l]; vr = xr[idx] + br[l]; }
  xl[idx] = vl;
  xr[idx] = vr;
  xlT[(b * L_ + l) * MP_ + m] = (bf16_t)vl;
}

// ---- attention logits e[b,i,j] with mask, stored transposed eT[b,j,i] -----
__global__ void k_e(const float* __restrict__ xl, const float* __restrict__ xr,
                    const float* __restrict__ att, const float* __restrict__ cg,
                    float* __restrict__ eT, int use_cg) {
  int idx = blockIdx.x;                     // b*M*M waves
  int j = idx % M_;
  int i = (idx / M_) % M_;
  int b = idx / (M_ * M_);
  int lane = threadIdx.x;
  const float* xli = xl + (b * MP_ + i) * L_;
  const float* xrj = xr + (b * MP_ + j) * L_;
  float s = 0.f;
  for (int l = lane; l < L_; l += 32) {
    float x = xli[l] + xrj[l];
    s += att[l] * (x > 0.f ? x : 0.2f * x);
  }
  for (int o = 16; o > 0; o >>= 1) s += __shfl_xor(s, o, 32);
  if (lane == 0) {
    bool msk = (i == j) || (i == N_) || (j == N_);
    if (!msk && use_cg) msk = cg[(b * N_ + i) * N_ + j] > 0.f;
    eT[(b * MP_ + j) * MP_ + i] = msk ? s : -1e30f;
  }
}

// ------- softmax over i for fixed (b,j); write bf16 alphaT[b,j,i] ----------
__global__ void k_softmax(const float* __restrict__ eT, bf16_t* __restrict__ alphaT) {
  int bj = blockIdx.x;                      // B*MP blocks
  int j = bj % MP_;
  int b = bj / MP_;
  int t = threadIdx.x;
  bf16_t* out = alphaT + (b * MP_ + j) * MP_;
  if (j >= M_) {
    if (t < MP_) out[t] = (bf16_t)0.f;
    return;
  }
  __shared__ float red[256];
  const float* row = eT + (b * MP_ + j) * MP_;
  float v = (t < M_) ? row[t] : -1e30f;
  red[t] = v; __syncthreads();
  for (int o = 128; o > 0; o >>= 1) { if (t < o) red[t] = fmaxf(red[t], red[t + o]); __syncthreads(); }
  float mx = red[0]; __syncthreads();
  float ex = (t < M_) ? __expf(v - mx) : 0.f;
  red[t] = ex; __syncthreads();
  for (int o = 128; o > 0; o >>= 1) { if (t < o) red[t] += red[t + o]; __syncthreads(); }
  float sum = red[0];
  if (t < MP_) out[t] = (bf16_t)((t < M_) ? ex / sum : 0.f);
}

// ---------- agg epilogue: relu(agg + gat_bias) -> bf16 for final GEMM ------
__global__ void k_agg_epi(const float* __restrict__ agg, const float* __restrict__ gbias,
                          bf16_t* __restrict__ aggrelu) {
  int idx = blockIdx.x * blockDim.x + threadIdx.x;
  if (idx >= B_ * MP_ * L_) return;
  int l = idx % L_;
  float v = agg[idx] + gbias[l];
  aggrelu[idx] = (bf16_t)(v > 0.f ? v : 0.f);
}

// --------- latent_y: out[:, :-1] + out_b, transposed to (B,C,H,W) ----------
__global__ void k_final(const float* __restrict__ trans0, const float* __restrict__ outb,
                        float* __restrict__ dout) {
  int idx = blockIdx.x * blockDim.x + threadIdx.x;
  if (idx >= B_ * C_ * N_) return;
  int n = idx % N_;
  int c = (idx / N_) % C_;
  int b = idx / (N_ * C_);
  dout[idx] = trans0[(b * MP_ + n) * C_ + c] + outb[c];
}

// --------------------------- ct_loss (single block) ------------------------
__global__ void k_loss(const float* __restrict__ cg, const float* __restrict__ latent,
                       const float* __restrict__ trans1, const float* __restrict__ outb,
                       float* __restrict__ dout) {
  __shared__ float red[256];
  int t = threadIdx.x;
  float s1 = 0.f;
  for (int idx = t; idx < B_ * N_ * N_; idx += 256) {
    int j = idx % N_;
    int i = (idx / N_) % N_;
    float d = ((i == j) ? 1.f : 0.f) - cg[idx];
    s1 += d * d;
  }
  float s2 = 0.f;
  for (int idx = t; idx < B_ * N_ * C_; idx += 256) {
    int c = idx % C_;
    int n = (idx / C_) % N_;
    int b = idx / (N_ * C_);
    float lat = latent[(b * C_ + c) * N_ + n];
    float rec = trans1[(b * MP_ + n) * C_ + c] + outb[c];
    float d = lat - rec;
    s2 += d * d;
  }
  red[t] = s1 / (float)(B_ * N_ * N_) + s2 / (float)(B_ * N_ * C_);
  __syncthreads();
  for (int o = 128; o > 0; o >>= 1) { if (t < o) red[t] += red[t + o]; __syncthreads(); }
  if (t == 0) dout[B_ * C_ * N_] = red[0];
}

// ===========================================================================
extern "C" void kernel_launch(void* const* d_in, const int* in_sizes, int n_in,
                              void* d_out, int out_size, void* d_ws, size_t ws_size,
                              hipStream_t stream) {
  const float* latent  = (const float*)d_in[0];
  const float* adb     = (const float*)d_in[2];
  const float* disc_w1 = (const float*)d_in[3];
  const float* disc_b1 = (const float*)d_in[4];
  const float* disc_w2 = (const float*)d_in[5];
  const float* disc_b2 = (const float*)d_in[6];
  const float* gat_wl  = (const float*)d_in[7];
  const float* gat_bl  = (const float*)d_in[8];
  const float* gat_wr  = (const float*)d_in[9];
  const float* gat_br  = (const float*)d_in[10];
  const float* gat_att = (const float*)d_in[11];
  const float* gat_bias= (const float*)d_in[12];
  const float* out_w   = (const float*)d_in[13];
  const float* out_b   = (const float*)d_in[14];
  float* dout = (float*)d_out;

  char* wsp = (char*)d_ws;
  size_t off = 0;
  auto alloc = [&](size_t bytes) -> char* {
    char* p = wsp + off;
    off = (off + bytes + 255) & ~(size_t)255;
    return p;
  };
  float*  pos_f    = (float*)alloc(sizeof(float) * B_ * N_ * C_);
  float*  pi_      = (float*)alloc(sizeof(float) * B_ * N_ * L_);
  float*  pj_      = (float*)alloc(sizeof(float) * B_ * N_ * L_);
  float*  pa_      = (float*)alloc(sizeof(float) * L_);
  float*  cg_      = (float*)alloc(sizeof(float) * B_ * N_ * N_);
  float*  nodes_f  = (float*)alloc(sizeof(float) * B_ * MP_ * C_);
  float*  xl_      = (float*)alloc(sizeof(float) * B_ * MP_ * L_);
  float*  xr_      = (float*)alloc(sizeof(float) * B_ * MP_ * L_);
  float*  eT_      = (float*)alloc(sizeof(float) * B_ * MP_ * MP_);
  float*  agg_     = (float*)alloc(sizeof(float) * B_ * MP_ * L_);
  float*  trans0   = (float*)alloc(sizeof(float) * B_ * MP_ * C_);
  float*  trans1   = (float*)alloc(sizeof(float) * B_ * MP_ * C_);
  bf16_t* pos_b    = (bf16_t*)alloc(sizeof(bf16_t) * B_ * N_ * C_);
  bf16_t* Wi_t     = (bf16_t*)alloc(sizeof(bf16_t) * L_ * C_);
  bf16_t* Wj_t     = (bf16_t*)alloc(sizeof(bf16_t) * L_ * C_);
  bf16_t* gwl_t    = (bf16_t*)alloc(sizeof(bf16_t) * L_ * C_);
  bf16_t* gwr_t    = (bf16_t*)alloc(sizeof(bf16_t) * L_ * C_);
  bf16_t* outw_t   = (bf16_t*)alloc(sizeof(bf16_t) * C_ * L_);
  bf16_t* nodes_b  = (bf16_t*)alloc(sizeof(bf16_t) * B_ * MP_ * C_);
  bf16_t* xlT_b    = (bf16_t*)alloc(sizeof(bf16_t) * B_ * L_ * MP_);
  bf16_t* alphaT_b = (bf16_t*)alloc(sizeof(bf16_t) * B_ * MP_ * MP_);
  bf16_t* aggrelu_b= (bf16_t*)alloc(sizeof(bf16_t) * B_ * MP_ * L_);

  auto gemm = [&](const bf16_t* A, int sA, const bf16_t* Bt, int sB,
                  float* Cm, int sC, int Mrows, int Ncols, int K,
                  int lda, int ldb, int ldc, int batch) {
    dim3 grid((Mrows / 16) * (Ncols / 16), batch);
    k_wmma_gemm<<<grid, 32, 0, stream>>>(A, sA, Bt, sB, Cm, sC, Ncols, K, lda, ldb, ldc);
  };
  auto gemm_lds = [&](const bf16_t* A, int sA, const bf16_t* Bt, int sB,
                      float* Cm, int sC, int Mrows, int Ncols, int K,
                      int lda, int ldb, int ldc, int batch) {
    dim3 grid((Mrows / 16) * (Ncols / 16), batch);
    k_wmma_gemm_lds<<<grid, 32, 0, stream>>>(A, sA, Bt, sB, Cm, sC, Ncols, K, lda, ldb, ldc);
  };

  // ---- stage 0: pos + PE, weight transposes (f32 -> bf16 N-major) ----
  k_pos<<<(B_ * N_ * C_ + 255) / 256, 256, 0, stream>>>(latent, pos_f, pos_b);
  k_transpose_bf16<<<(C_ * L_ + 255) / 256, 256, 0, stream>>>(disc_w1,            Wi_t,  C_, L_);
  k_transpose_bf16<<<(C_ * L_ + 255) / 256, 256, 0, stream>>>(disc_w1 + C_ * L_,  Wj_t,  C_, L_);
  k_transpose_bf16<<<(C_ * L_ + 255) / 256, 256, 0, stream>>>(gat_wl,             gwl_t, C_, L_);
  k_transpose_bf16<<<(C_ * L_ + 255) / 256, 256, 0, stream>>>(gat_wr,             gwr_t, C_, L_);
  k_transpose_bf16<<<(C_ * L_ + 255) / 256, 256, 0, stream>>>(out_w,              outw_t, L_, C_);
  k_pa<<<(L_ + 127) / 128, 128, 0, stream>>>(adb, disc_w1 + 2 * C_ * L_, disc_b1, pa_);

  // ---- stage 1: discriminator -> binary causal graph cg ----
  gemm(pos_b, 0, Wi_t, 0, pi_, 0, B_ * N_, L_, C_, C_, C_, L_, 1);
  gemm(pos_b, 0, Wj_t, 0, pj_, 0, B_ * N_, L_, C_, C_, C_, L_, 1);
  k_cg<<<B_ * N_ * N_, 32, 0, stream>>>(pi_, pj_, pa_, disc_w2, disc_b2, cg_);

  // ---- stage 2: two GAT-transitioner passes (cg-adjacency, identity) ----
  auto run_pass = [&](unsigned salt, int use_cg, float* trans) {
    k_nodes<<<(B_ * MP_ * C_ + 255) / 256, 256, 0, stream>>>(pos_f, adb, nodes_f, nodes_b, salt);
    gemm_lds(nodes_b, 0, gwl_t, 0, xl_, 0, B_ * MP_, L_, C_, C_, C_, L_, 1);
    gemm_lds(nodes_b, 0, gwr_t, 0, xr_, 0, B_ * MP_, L_, C_, C_, C_, L_, 1);
    k_xlxr_epi<<<(B_ * MP_ * L_ + 255) / 256, 256, 0, stream>>>(xl_, xr_, gat_bl, gat_br, xlT_b);
    k_e<<<B_ * M_ * M_, 32, 0, stream>>>(xl_, xr_, gat_att, cg_, eT_, use_cg);
    k_softmax<<<B_ * MP_, 256, 0, stream>>>(eT_, alphaT_b);
    gemm_lds(alphaT_b, MP_ * MP_, xlT_b, L_ * MP_, agg_, MP_ * L_,
             MP_, L_, MP_, MP_, MP_, L_, B_);
    k_agg_epi<<<(B_ * MP_ * L_ + 255) / 256, 256, 0, stream>>>(agg_, gat_bias, aggrelu_b);
    gemm_lds(aggrelu_b, 0, outw_t, 0, trans, 0, B_ * MP_, C_, L_, L_, L_, C_, 1);
  };
  run_pass(0x1111u, /*use_cg=*/1, trans0);   // -> latent_y
  run_pass(0x2222u, /*use_cg=*/0, trans1);   // -> recon

  // ---- stage 3: outputs ----
  k_final<<<(B_ * C_ * N_ + 255) / 256, 256, 0, stream>>>(trans0, out_b, dout);
  k_loss<<<1, 256, 0, stream>>>(cg_, latent, trans1, out_b, dout);
}
